// MultiHeadAttention_60567628808747
// MI455X (gfx1250) — matrix operand order repaired and checked by reading, not compiled, not run on hardware
//
#include <hip/hip_runtime.h>

typedef _Float16 h16;
typedef __attribute__((ext_vector_type(16))) _Float16 v16h;
typedef __attribute__((ext_vector_type(8)))  _Float16 v8h;
typedef __attribute__((ext_vector_type(8)))  float    v8f;
typedef __attribute__((ext_vector_type(4)))  int      v4i;

namespace {
constexpr int kB = 2, kS = 2048, kD = 1024, kH = 16, kDK = 64;
constexpr int kM = kB * kS;            // 4096 flattened rows
constexpr float kScale = 0.125f;       // 1/sqrt(DK)
}

// ---- gfx1250 async global->LDS path (guarded; falls back to sync copy) ----
#if defined(__gfx1250__) &&                                                   \
    __has_builtin(__builtin_amdgcn_global_load_async_to_lds_b128) &&          \
    __has_builtin(__builtin_amdgcn_s_wait_asynccnt)
#define USE_ASYNC_LDS 1
#else
#define USE_ASYNC_LDS 0
#endif

__device__ __forceinline__ void lds_copy16(h16* lds_dst, const h16* gsrc) {
#if USE_ASYNC_LDS
  __builtin_amdgcn_global_load_async_to_lds_b128(
      (__attribute__((address_space(1))) v4i*)gsrc,
      (__attribute__((address_space(3))) v4i*)lds_dst, 0, 0);
#else
  *(v8h*)lds_dst = *(const v8h*)gsrc;
#endif
}

__device__ __forceinline__ void lds_copy_wait() {
#if USE_ASYNC_LDS
  __builtin_amdgcn_s_wait_asynccnt(0);
#endif
}

__device__ __forceinline__ v8f wmma16(v16h a, v16h b, v8f c) {
  // D = A(16x32 f16) * B(32x16 f16) + C(16x16 f32)
  return __builtin_amdgcn_wmma_f32_16x16x32_f16(false, a, false, b, (short)0, c,
                                                false, false);
}

// A fragment (16x32, MxK) from a row-major [row][k] LDS tile.
__device__ __forceinline__ v16h frag_a(const h16* tile, int pitch, int row0,
                                       int kbase, int lane) {
  const int r  = row0 + (lane & 15);
  const int ko = kbase + ((lane & 16) ? 8 : 0);
  const h16* p = tile + r * pitch + ko;
  union { v16h v; v8h h[2]; } u;
  u.h[0] = *(const v8h*)(p);
  u.h[1] = *(const v8h*)(p + 16);
  return u.v;
}

// B fragment (32x16, KxN) sourced from a row-major [n][k] LDS tile (B^T rows).
__device__ __forceinline__ v16h frag_b(const h16* tile, int pitch, int n0,
                                       int kbase, int lane) {
  const int n  = n0 + (lane & 15);
  const int ko = kbase + ((lane & 16) ? 8 : 0);
  const h16* p = tile + n * pitch + ko;
  union { v16h v; v8h h[2]; } u;
  u.h[0] = *(const v8h*)(p);
  u.h[1] = *(const v8h*)(p + 16);
  return u.v;
}

// out = X(f32,[M,D]) @ W(f32,[D,D])^T + bias, f16 into [B,H,S,DK].
// Block tile 64x128, 8 waves, each wave 32x32 (4 WMMA / step),
// ping-pong LDS + register prefetch: 1 barrier per K-step.
__global__ __launch_bounds__(256) void proj_qkv_kernel(
    const float* __restrict__ X, const float* __restrict__ W,
    const float* __restrict__ bias, h16* __restrict__ dst) {
  __shared__ h16 Xs[2][64 * 40];
  __shared__ h16 Ws[2][128 * 40];
  const int m0 = blockIdx.x * 64;
  const int n0 = blockIdx.y * 128;
  const int t = threadIdx.x, lane = t & 31, wave = t >> 5;
  const int r0l = (wave & 1) * 32;    // wave rows within tile
  const int n0l = (wave >> 1) * 32;   // wave cols within tile
  const int sr = t >> 2;              // staging row 0..63
  const int sc = (t & 3) * 8;         // staging col 0,8,16,24

  float xr[8], wr0[8], wr1[8];
  auto load_regs = [&](int k0) {
    const float* xp  = X + (size_t)(m0 + sr) * kD + k0 + sc;
    const float* wp0 = W + (size_t)(n0 + sr) * kD + k0 + sc;
    const float* wp1 = W + (size_t)(n0 + 64 + sr) * kD + k0 + sc;
#pragma unroll
    for (int i = 0; i < 8; ++i) { xr[i] = xp[i]; wr0[i] = wp0[i]; wr1[i] = wp1[i]; }
  };
  auto store_regs = [&](int buf) {
#pragma unroll
    for (int i = 0; i < 8; ++i) {
      Xs[buf][sr * 40 + sc + i]        = (h16)xr[i];
      Ws[buf][sr * 40 + sc + i]        = (h16)wr0[i];
      Ws[buf][(sr + 64) * 40 + sc + i] = (h16)wr1[i];
    }
  };

  v8f c00 = {}, c01 = {}, c10 = {}, c11 = {};
  load_regs(0);
  store_regs(0);
  __syncthreads();
  int cur = 0;
  for (int it = 0; it < kD / 32; ++it) {
    const bool more = (it + 1) < kD / 32;
    if (more) load_regs((it + 1) * 32);   // overlap global latency with WMMA
    v16h a0 = frag_a(&Xs[cur][0], 40, r0l, 0, lane);
    v16h a1 = frag_a(&Xs[cur][0], 40, r0l + 16, 0, lane);
    v16h b0 = frag_b(&Ws[cur][0], 40, n0l, 0, lane);
    v16h b1 = frag_b(&Ws[cur][0], 40, n0l + 16, 0, lane);
    c00 = wmma16(a0, b0, c00);
    c01 = wmma16(a0, b1, c01);
    c10 = wmma16(a1, b0, c10);
    c11 = wmma16(a1, b1, c11);
    if (more) {
      const int nxt = cur ^ 1;
      store_regs(nxt);
      __syncthreads();
      cur = nxt;
    }
  }

  const int half = (lane >> 4) & 1, ln = lane & 15;
  const int nnA = n0 + n0l + ln, nnB = nnA + 16;
  const float bbA = bias[nnA], bbB = bias[nnB];
  auto put = [&](int mm, int nn, float val) {
    const int b = mm >> 11, s = mm & (kS - 1);
    const int hh = nn >> 6, dk = nn & (kDK - 1);
    dst[(((size_t)b * kH + hh) * kS + s) * kDK + dk] = (h16)val;
  };
#pragma unroll
  for (int v = 0; v < 8; ++v) {
    const int mm = m0 + r0l + v + 8 * half;
    put(mm,      nnA, c00[v] + bbA);
    put(mm,      nnB, c01[v] + bbB);
    put(mm + 16, nnA, c10[v] + bbA);
    put(mm + 16, nnB, c11[v] + bbB);
  }
}

// Flash attention: grid (S/64, B*H), 128 threads = 4 waves, wave owns 16 q rows.
// K tiles stream via async global->LDS; V tiles via register prefetch (transposed
// store); ping-pong LDS: 1 barrier per KV-step.
__global__ __launch_bounds__(128) void flash_attn_kernel(
    const h16* __restrict__ Qh, const h16* __restrict__ Kh,
    const h16* __restrict__ Vh, const int* __restrict__ mask,
    h16* __restrict__ Oc) {
  __shared__ h16 Qs[64 * 72];
  __shared__ h16 Ks[2][64 * 72];
  __shared__ h16 Vt[2][64 * 72];     // V transposed: [d][j]
  __shared__ h16 Ps[4 * 16 * 72];    // per-wave P bounce buffer
  const int t = threadIdx.x, lane = t & 31, wave = t >> 5;
  const int half = (lane >> 4) & 1;
  const int bh = blockIdx.y;
  const int b = bh / kH, h = bh % kH;
  const int q0 = blockIdx.x * 64;
  const h16* Qb = Qh + (size_t)bh * kS * kDK;
  const h16* Kb = Kh + (size_t)bh * kS * kDK;
  const h16* Vb = Vh + (size_t)bh * kS * kDK;

  const int srow = t >> 1, scol = (t & 1) * 32;   // staging coords

  auto issue_k = [&](int buf, int kv0) {
    const h16* src = Kb + (size_t)(kv0 + srow) * kDK + scol;
#pragma unroll
    for (int i = 0; i < 4; ++i)
      lds_copy16(&Ks[buf][srow * 72 + scol + i * 8], src + i * 8);
  };

  v8h vr[4];
  auto load_v = [&](int kv0) {
    const h16* src = Vb + (size_t)(kv0 + srow) * kDK + scol;
#pragma unroll
    for (int i = 0; i < 4; ++i) vr[i] = *(const v8h*)(src + i * 8);
  };
  auto store_vt = [&](int buf) {
#pragma unroll
    for (int i = 0; i < 4; ++i)
#pragma unroll
      for (int e = 0; e < 8; ++e)
        Vt[buf][(scol + i * 8 + e) * 72 + srow] = vr[i][e];
  };

  {  // stage Q tile once (async) + prologue of the KV pipeline
    const h16* qsrc = Qb + (size_t)(q0 + srow) * kDK + scol;
#pragma unroll
    for (int i = 0; i < 4; ++i)
      lds_copy16(&Qs[srow * 72 + scol + i * 8], qsrc + i * 8);
    issue_k(0, 0);
    load_v(0);
    store_vt(0);
    lds_copy_wait();
  }
  __syncthreads();

  const int r0 = wave * 16;
  v16h aq0 = frag_a(Qs, 72, r0, 0, lane);
  v16h aq1 = frag_a(Qs, 72, r0, 32, lane);

  v8f o0 = {}, o1 = {}, o2 = {}, o3 = {};
  float rmax[8], rsum[8];
#pragma unroll
  for (int v = 0; v < 8; ++v) { rmax[v] = -1e30f; rsum[v] = 0.f; }
  h16* Pw = Ps + wave * 16 * 72;

  int cur = 0;
  for (int it = 0; it < kS / 64; ++it) {
    const int kv0 = it * 64;
    const bool more = (it + 1) < kS / 64;
    if (more) {                      // prefetch next tile while computing
      issue_k(cur ^ 1, kv0 + 64);
      load_v(kv0 + 64);
    }

    // scores S = Q K^T (16 rows x 64 cols), scale + mask
    v8f sf[4];
#pragma unroll
    for (int f = 0; f < 4; ++f) {
      v16h b0 = frag_b(&Ks[cur][0], 72, 16 * f, 0, lane);
      v16h b1 = frag_b(&Ks[cur][0], 72, 16 * f, 32, lane);
      v8f s = {};
      s = wmma16(aq0, b0, s);
      s = wmma16(aq1, b1, s);
      const int jj = kv0 + 16 * f + (lane & 15);
#pragma unroll
      for (int v = 0; v < 8; ++v) {
        const int qi = q0 + r0 + v + 8 * half;
        const int mv = mask[(size_t)qi * kS + jj];
        s[v] = (mv == 0) ? -1e9f : s[v] * kScale;
      }
      sf[f] = s;
    }

    // online softmax (16-lane halves hold disjoint rows)
#pragma unroll
    for (int v = 0; v < 8; ++v) {
      float m = fmaxf(fmaxf(sf[0][v], sf[1][v]), fmaxf(sf[2][v], sf[3][v]));
#pragma unroll
      for (int x = 1; x < 16; x <<= 1) m = fmaxf(m, __shfl_xor(m, x, 16));
      m = fmaxf(m, rmax[v]);
      const float corr = __expf(rmax[v] - m);
      rmax[v] = m;
      float ps = 0.f;
#pragma unroll
      for (int f = 0; f < 4; ++f) {
        const float e = __expf(sf[f][v] - m);
        sf[f][v] = e;
        ps += e;
      }
#pragma unroll
      for (int x = 1; x < 16; x <<= 1) ps += __shfl_xor(ps, x, 16);
      rsum[v] = rsum[v] * corr + ps;
      o0[v] *= corr; o1[v] *= corr; o2[v] *= corr; o3[v] *= corr;
    }

    // C-layout -> A-layout via per-wave LDS bounce (same-wave DS ops in order)
#pragma unroll
    for (int f = 0; f < 4; ++f)
#pragma unroll
      for (int v = 0; v < 8; ++v)
        Pw[(v + 8 * half) * 72 + 16 * f + (lane & 15)] = (h16)sf[f][v];

    v16h ap0 = frag_a(Pw, 72, 0, 0, lane);
    v16h ap1 = frag_a(Pw, 72, 0, 32, lane);
    const h16* vt = &Vt[cur][0];
    o0 = wmma16(ap0, frag_b(vt, 72, 0, 0, lane), o0);
    o0 = wmma16(ap1, frag_b(vt, 72, 0, 32, lane), o0);
    o1 = wmma16(ap0, frag_b(vt, 72, 16, 0, lane), o1);
    o1 = wmma16(ap1, frag_b(vt, 72, 16, 32, lane), o1);
    o2 = wmma16(ap0, frag_b(vt, 72, 32, 0, lane), o2);
    o2 = wmma16(ap1, frag_b(vt, 72, 32, 32, lane), o2);
    o3 = wmma16(ap0, frag_b(vt, 72, 48, 0, lane), o3);
    o3 = wmma16(ap1, frag_b(vt, 72, 48, 32, lane), o3);

    if (more) {
      const int nxt = cur ^ 1;
      store_vt(nxt);
      lds_copy_wait();
      __syncthreads();
      cur = nxt;
    }
  }

  // normalize, store f16 into concat layout [b][s][h*DK + d]
#pragma unroll
  for (int v = 0; v < 8; ++v) {
    const float inv = 1.f / rsum[v];
    const int m = q0 + r0 + v + 8 * half;
    const size_t base = ((size_t)b * kS + m) * kD + (size_t)h * kDK + (lane & 15);
    Oc[base +  0] = (h16)(o0[v] * inv);
    Oc[base + 16] = (h16)(o1[v] * inv);
    Oc[base + 32] = (h16)(o2[v] * inv);
    Oc[base + 48] = (h16)(o3[v] * inv);
  }
}

// out = concat(f16,[M,D]) @ Wo^T + bo -> f32 d_out. Same 64x128 tiling.
__global__ __launch_bounds__(256) void out_proj_kernel(
    const h16* __restrict__ Xh, const float* __restrict__ W,
    const float* __restrict__ bias, float* __restrict__ out) {
  __shared__ h16 Xs[2][64 * 40];
  __shared__ h16 Ws[2][128 * 40];
  const int m0 = blockIdx.x * 64;
  const int n0 = blockIdx.y * 128;
  const int t = threadIdx.x, lane = t & 31, wave = t >> 5;
  const int r0l = (wave & 1) * 32, n0l = (wave >> 1) * 32;
  const int sr = t >> 2, sc = (t & 3) * 8;

  v8h xr;
  float wr0[8], wr1[8];
  auto load_regs = [&](int k0) {
    const h16* xp    = Xh + (size_t)(m0 + sr) * kD + k0 + sc;
    const float* wp0 = W + (size_t)(n0 + sr) * kD + k0 + sc;
    const float* wp1 = W + (size_t)(n0 + 64 + sr) * kD + k0 + sc;
    xr = *(const v8h*)xp;
#pragma unroll
    for (int i = 0; i < 8; ++i) { wr0[i] = wp0[i]; wr1[i] = wp1[i]; }
  };
  auto store_regs = [&](int buf) {
    *(v8h*)&Xs[buf][sr * 40 + sc] = xr;
#pragma unroll
    for (int i = 0; i < 8; ++i) {
      Ws[buf][sr * 40 + sc + i]        = (h16)wr0[i];
      Ws[buf][(sr + 64) * 40 + sc + i] = (h16)wr1[i];
    }
  };

  v8f c00 = {}, c01 = {}, c10 = {}, c11 = {};
  load_regs(0);
  store_regs(0);
  __syncthreads();
  int cur = 0;
  for (int it = 0; it < kD / 32; ++it) {
    const bool more = (it + 1) < kD / 32;
    if (more) load_regs((it + 1) * 32);
    v16h a0 = frag_a(&Xs[cur][0], 40, r0l, 0, lane);
    v16h a1 = frag_a(&Xs[cur][0], 40, r0l + 16, 0, lane);
    v16h b0 = frag_b(&Ws[cur][0], 40, n0l, 0, lane);
    v16h b1 = frag_b(&Ws[cur][0], 40, n0l + 16, 0, lane);
    c00 = wmma16(a0, b0, c00);
    c01 = wmma16(a0, b1, c01);
    c10 = wmma16(a1, b0, c10);
    c11 = wmma16(a1, b1, c11);
    if (more) {
      const int nxt = cur ^ 1;
      store_regs(nxt);
      __syncthreads();
      cur = nxt;
    }
  }

  const int half = (lane >> 4) & 1, ln = lane & 15;
  const int nnA = n0 + n0l + ln, nnB = nnA + 16;
  const float bbA = bias[nnA], bbB = bias[nnB];
#pragma unroll
  for (int v = 0; v < 8; ++v) {
    const int mm = m0 + r0l + v + 8 * half;
    out[(size_t)mm * kD + nnA]        = c00[v] + bbA;
    out[(size_t)mm * kD + nnB]        = c01[v] + bbB;
    out[(size_t)(mm + 16) * kD + nnA] = c10[v] + bbA;
    out[(size_t)(mm + 16) * kD + nnB] = c11[v] + bbB;
  }
}

extern "C" void kernel_launch(void* const* d_in, const int* in_sizes, int n_in,
                              void* d_out, int out_size, void* d_ws, size_t ws_size,
                              hipStream_t stream) {
  (void)in_sizes; (void)n_in; (void)out_size; (void)ws_size;
  const float* q  = (const float*)d_in[0];
  const float* k  = (const float*)d_in[1];
  const float* v  = (const float*)d_in[2];
  const int* mask = (const int*)d_in[3];
  const float* Wq = (const float*)d_in[4];
  const float* bq = (const float*)d_in[5];
  const float* Wk = (const float*)d_in[6];
  const float* bk = (const float*)d_in[7];
  const float* Wv = (const float*)d_in[8];
  const float* bv = (const float*)d_in[9];
  const float* Wo = (const float*)d_in[10];
  const float* bo = (const float*)d_in[11];

  // workspace: 3 head-split f16 buffers (8 MB each) + concat f16 (8 MB) = 32 MB
  h16* qh = (h16*)d_ws;
  const size_t perBuf = (size_t)kB * kH * kS * kDK;  // 4M halves
  h16* kh = qh + perBuf;
  h16* vh = kh + perBuf;
  h16* oc = vh + perBuf;

  dim3 gp(kM / 64, kD / 128);
  proj_qkv_kernel<<<gp, 256, 0, stream>>>(q, Wq, bq, qh);
  proj_qkv_kernel<<<gp, 256, 0, stream>>>(k, Wk, bk, kh);
  proj_qkv_kernel<<<gp, 256, 0, stream>>>(v, Wv, bv, vh);

  dim3 ga(kS / 64, kB * kH);
  flash_attn_kernel<<<ga, 128, 0, stream>>>(qh, kh, vh, mask, oc);

  out_proj_kernel<<<gp, 256, 0, stream>>>(oc, Wo, bo, (float*)d_out);
}
